// AutoCorrelation_42717744726702
// MI455X (gfx1250) — matrix operand order repaired and checked by reading, not compiled, not run on hardware
//
#include <hip/hip_runtime.h>
#include <hip/hip_bf16.h>
#include <math.h>

// ---------------------------------------------------------------------------
// AutoCorrelation (Autoformer-style) for MI455X / gfx1250, wave32 + WMMA.
//
// Pipeline (all matrix work through v_wmma_f32_16x16x32_bf16):
//   1) qp/kp/vp projections: [8192x1024]x[1024x1024] bf16 WMMA GEMM (+bias).
//      qp/kp stored TRANSPOSED as [B*D, L] so the DFT GEMMs are row-major.
//   2) rfft via DFT-as-GEMM: F = qpT @ [cos | -sin]  ([2048 x 2176] basis,
//      1025 bins padded to 1088; Re and Im concatenated along N).
//   3) spectral product P = Fq * conj(Fk)  (elementwise, bf16).
//   4) irfft via GEMM: corr = [P_re | P_im] @ [scale*cos ; -scale*sin]
//      with a fused epilogue: max over tau in [1,2047] via atomic float-max
//      (corr tensor is never written to memory).
//   5) softmax over the 64 d_k channels per (b,h); shift = (int)weight.
//   6) roll vp by shift (0 or 1), final GEMM @ Wo + bo -> fp32 out.
//
// CDNA5-specific paths: v_wmma_f32_16x16x32_bf16 for all matmuls;
// global_load_async_to_lds_b128 (+ s_wait_asynccnt) for A-tile staging in the
// bf16-A GEMMs; global_prefetch_b8 hints in the loaders.
//
// Workspace requirement: ~150 MB (see layout in kernel_launch).
// ---------------------------------------------------------------------------

#define LSEQ   2048
#define DMODEL 1024
#define BATCH  4
#define NHEAD  16
#define NDK    64
#define FPAD   1088          // 1025 rfft bins padded to multiple of 64
#define NF2    (2*FPAD)      // 2176 = Re|Im concat, multiple of 128
#define MROWS  (BATCH*LSEQ)  // 8192
#define NCH    (BATCH*DMODEL)// 4096 DFT channels

typedef __attribute__((ext_vector_type(16))) __bf16 v16bf;
typedef __attribute__((ext_vector_type(8)))  float  v8f;
typedef __attribute__((ext_vector_type(4)))  unsigned int uint4v;
typedef __attribute__((ext_vector_type(4)))  float  float4v;

union FragBF { uint4v u[2]; v16bf v; };

__device__ __forceinline__ unsigned short f2bf(float f) {
    unsigned u = __float_as_uint(f);
    unsigned r = u + 0x7FFFu + ((u >> 16) & 1u);   // round-to-nearest-even
    return (unsigned short)(r >> 16);
}
__device__ __forceinline__ float bf2f(unsigned short h) {
    return __uint_as_float(((unsigned)h) << 16);
}
__device__ __forceinline__ void atomicMaxF32(float* addr, float val) {
    // Sign-split monotonic trick: correct for mixed-sign floats, addr init -inf.
    if (val >= 0.0f) atomicMax((int*)addr, __float_as_int(val));
    else             atomicMin((unsigned int*)addr, __float_as_uint(val));
}

// ---------------------------------------------------------------------------
// Generic bf16 WMMA GEMM: C[M,N] = A[M,K] * B[K,N] (+bias)
//   AMODE: 0 = A is fp32 row-major (convert to bf16 on load, VGPR staging)
//          1 = A is bf16 row-major (async DMA: global_load_async_to_lds_b128)
//   STORE: 0 = fp32 row-major        1 = bf16 row-major
//          2 = bf16 transposed-per-batch: out[(m/LSEQ)*N + n][m%LSEQ]
//          3 = atomic row-max into float C[M], skipping column n==0 (tau=0)
// Block: 256 threads (8 waves), tile 128x128, K-step 32, waves 2(M)x4(N),
// each wave 64x32 = 8 WMMA accumulators.
// ---------------------------------------------------------------------------
template<int AMODE, int STORE, bool BIAS>
__global__ __launch_bounds__(256) void k_gemm(
    const void* __restrict__ Ap, const unsigned short* __restrict__ Bp,
    const float* __restrict__ bias, void* __restrict__ Cp,
    int M, int N, int K)
{
    __shared__ unsigned short As[128 * 40];   // padded stride 40 (80B, 16B-aligned)
    __shared__ unsigned short Bs[128 * 40];   // stored transposed: [n][k]

    const int tid   = threadIdx.x;
    const int lane  = tid & 31;
    const int wave  = tid >> 5;
    const int waveM = wave & 1;               // 2 waves along M (64 rows each)
    const int waveN = wave >> 1;              // 4 waves along N (32 cols each)
    const int half  = lane >> 4;
    const int l16   = lane & 15;
    const int rowBase = blockIdx.y * 128;
    const int colBase = blockIdx.x * 128;

    const int ar = tid >> 1;                  // A stage: row 0..127
    const int ac = (tid & 1) * 16;            // A stage: col 0 or 16
    const int br = tid >> 3;                  // B stage: k-row 0..31
    const int bc = (tid & 7) * 16;            // B stage: col chunk

    v8f acc[4][2] = {};

    for (int k0 = 0; k0 < K; k0 += 32) {
        // ---- stage A tile (128 x 32) into LDS ----
        if (AMODE == 0) {
            const float* A = (const float*)Ap;
            const float* src = A + (size_t)(rowBase + ar) * K + k0 + ac;
            __builtin_prefetch(src + 32, 0, 1);     // global_prefetch_b8
            unsigned short tmp[16];
            #pragma unroll
            for (int i = 0; i < 16; i += 4) {
                float4v f = *(const float4v*)(src + i);
                tmp[i+0] = f2bf(f.x); tmp[i+1] = f2bf(f.y);
                tmp[i+2] = f2bf(f.z); tmp[i+3] = f2bf(f.w);
            }
            *(uint4v*)&As[ar*40 + ac]     = *(const uint4v*)&tmp[0];
            *(uint4v*)&As[ar*40 + ac + 8] = *(const uint4v*)&tmp[8];
        } else {
            // Async DMA: 32 contiguous bytes global -> LDS, tracked by ASYNCcnt.
            // the instruction offset is added to both the LDS and the global address,
            // so the same base pair serves both 16B halves via offset:16.
            const unsigned short* A = (const unsigned short*)Ap;
            const unsigned short* src = A + (size_t)(rowBase + ar) * K + k0 + ac;
            unsigned lds0 = (unsigned)(uintptr_t)(&As[ar*40 + ac]); // LDS byte offset
            unsigned long long g0 = (unsigned long long)(uintptr_t)src;
            asm volatile("global_load_async_to_lds_b128 %0, %1, off"
                         :: "v"(lds0), "v"(g0) : "memory");
            asm volatile("global_load_async_to_lds_b128 %0, %1, off offset:16"
                         :: "v"(lds0), "v"(g0) : "memory");
        }
        // ---- stage B tile (32 x 128), transposed into LDS as [n][k] ----
        {
            const unsigned short* src = Bp + (size_t)(k0 + br) * N + colBase + bc;
            __builtin_prefetch(src + (size_t)32 * N, 0, 1);
            unsigned short tmpb[16];
            *(uint4v*)&tmpb[0] = *(const uint4v*)(src);
            *(uint4v*)&tmpb[8] = *(const uint4v*)(src + 8);
            #pragma unroll
            for (int i = 0; i < 16; i++) Bs[(bc + i) * 40 + br] = tmpb[i];
        }
        if (AMODE == 1)
            asm volatile("s_wait_asynccnt 0x0" ::: "memory");
        __syncthreads();

        // ---- 8x WMMA per wave per K-step ----
        FragBF bfrag[2];
        #pragma unroll
        for (int tn = 0; tn < 2; tn++) {
            int n = waveN * 32 + tn * 16 + l16;
            // B (32x16 bf16): lane = col n; elems 0-7 -> K=8*half.., 8-15 -> K=16+8*half..
            bfrag[tn].u[0] = *(const uint4v*)&Bs[n*40 + half*8];
            bfrag[tn].u[1] = *(const uint4v*)&Bs[n*40 + 16 + half*8];
        }
        #pragma unroll
        for (int tm = 0; tm < 4; tm++) {
            FragBF a;
            int r = waveM * 64 + tm * 16 + l16;
            // A (16x32 bf16): lane = row; elems 0-7 -> K=8*half.., 8-15 -> K=16+8*half..
            a.u[0] = *(const uint4v*)&As[r*40 + half*8];
            a.u[1] = *(const uint4v*)&As[r*40 + 16 + half*8];
            #pragma unroll
            for (int tn = 0; tn < 2; tn++) {
                acc[tm][tn] = __builtin_amdgcn_wmma_f32_16x16x32_bf16(
                    false, a.v, false, bfrag[tn].v,
                    (short)0, acc[tm][tn], false, false);
            }
        }
        __syncthreads();
    }

    // ---- epilogue ----
    if (STORE == 3) {
        // fused row-max over tau (N axis), excluding tau==0
        #pragma unroll
        for (int tm = 0; tm < 4; tm++) {
            #pragma unroll
            for (int v = 0; v < 8; v++) {
                int n0 = colBase + waveN * 32 + l16;      // tn=0 column
                float x0 = acc[tm][0][v];
                if (n0 == 0) x0 = -INFINITY;              // exclude lag 0
                float x = fmaxf(x0, acc[tm][1][v]);
                #pragma unroll
                for (int off = 1; off < 16; off <<= 1)
                    x = fmaxf(x, __shfl_xor(x, off, 32));
                if (l16 == 0) {
                    int m = rowBase + waveM * 64 + tm * 16 + v + 8 * half;
                    atomicMaxF32(((float*)Cp) + m, x);
                }
            }
        }
    } else {
        #pragma unroll
        for (int tm = 0; tm < 4; tm++) {
            #pragma unroll
            for (int tn = 0; tn < 2; tn++) {
                int n = colBase + waveN * 32 + tn * 16 + l16;
                float bv_ = 0.0f;
                if (BIAS) bv_ = bias[n];
                #pragma unroll
                for (int v = 0; v < 8; v++) {
                    int m = rowBase + waveM * 64 + tm * 16 + v + 8 * half;
                    float val = acc[tm][tn][v] + bv_;
                    if (STORE == 0) {
                        ((float*)Cp)[(size_t)m * N + n] = val;
                    } else if (STORE == 1) {
                        ((unsigned short*)Cp)[(size_t)m * N + n] = f2bf(val);
                    } else { // STORE == 2: transposed per batch of LSEQ rows
                        int b = m >> 11, t = m & (LSEQ - 1);  // LSEQ == 2048
                        ((unsigned short*)Cp)[((size_t)b * N + n) * LSEQ + t] = f2bf(val);
                    }
                }
            }
        }
    }
}

// ---------------------------------------------------------------------------
// Small helper kernels
// ---------------------------------------------------------------------------
__global__ void k_f32_to_bf16(const float* __restrict__ src,
                              unsigned short* __restrict__ dst, int n) {
    int i = blockIdx.x * blockDim.x + threadIdx.x;
    if (i < n) dst[i] = f2bf(src[i]);
}

// Forward DFT basis [LSEQ x NF2]: cols [0,FPAD)= cos(2pi t f/L) (f<=1024),
// cols [FPAD,NF2) = -sin(2pi t f/L); padded bins are zero.
__global__ void k_basis_fwd(unsigned short* __restrict__ Bf) {
    int idx = blockIdx.x * blockDim.x + threadIdx.x;
    int t = idx / NF2, j = idx % NF2;
    int f = (j < FPAD) ? j : j - FPAD;
    float val = 0.0f;
    if (f <= LSEQ / 2) {
        double th = 6.283185307179586 * (double)((t * f) & (LSEQ - 1)) / (double)LSEQ;
        val = (j < FPAD) ? (float)cos(th) : (float)(-sin(th));
    }
    Bf[idx] = f2bf(val);
}

// Inverse DFT basis [NF2 x LSEQ]: rows [0,FPAD)= s_f*cos(2pi f tau/L),
// rows [FPAD,NF2) = -s_f*sin(2pi f tau/L); s_f = 1/L at f=0,1024 else 2/L.
__global__ void k_basis_inv(unsigned short* __restrict__ Bi) {
    int idx = blockIdx.x * blockDim.x + threadIdx.x;
    int kk = idx / LSEQ, tau = idx % LSEQ;
    int f = (kk < FPAD) ? kk : kk - FPAD;
    float val = 0.0f;
    if (f <= LSEQ / 2) {
        double th = 6.283185307179586 * (double)((f * tau) & (LSEQ - 1)) / (double)LSEQ;
        float s = (f == 0 || f == LSEQ / 2) ? (1.0f / LSEQ) : (2.0f / LSEQ);
        val = (kk < FPAD) ? s * (float)cos(th) : -s * (float)sin(th);
    }
    Bi[idx] = f2bf(val);
}

// P = Fq * conj(Fk): [NCH x NF2] with Re in [0,FPAD), Im in [FPAD,NF2).
__global__ void k_spectrum(const unsigned short* __restrict__ Fq,
                           const unsigned short* __restrict__ Fk,
                           unsigned short* __restrict__ P) {
    int idx = blockIdx.x * blockDim.x + threadIdx.x;
    int ch = idx / NF2, j = idx % NF2;
    int f = (j < FPAD) ? j : j - FPAD;
    if (f > LSEQ / 2) { P[idx] = 0; return; }
    size_t base = (size_t)ch * NF2;
    float qr = bf2f(Fq[base + f]),        qi = bf2f(Fq[base + FPAD + f]);
    float kr = bf2f(Fk[base + f]),        ki = bf2f(Fk[base + FPAD + f]);
    float val = (j < FPAD) ? (qr * kr + qi * ki)    // Re(q * conj(k))
                           : (qi * kr - qr * ki);   // Im(q * conj(k))
    P[idx] = f2bf(val);
}

__global__ void k_init_max(float* m, int n) {
    int i = blockIdx.x * blockDim.x + threadIdx.x;
    if (i < n) m[i] = -INFINITY;
}

// Per (b,h): softmax of top-1 corr over the 64 d_k channels; shift = (int)w.
__global__ void k_shift(const float* __restrict__ m, int* __restrict__ shift) {
    __shared__ float sm[64];
    int bh = blockIdx.x;      // b*NHEAD + h  -> channel base bh*64
    int c  = threadIdx.x;
    float v = m[bh * 64 + c];
    sm[c] = v; __syncthreads();
    for (int s = 32; s > 0; s >>= 1) { if (c < s) sm[c] = fmaxf(sm[c], sm[c + s]); __syncthreads(); }
    float mx = sm[0]; __syncthreads();
    float e = __expf(v - mx);
    sm[c] = e; __syncthreads();
    for (int s = 32; s > 0; s >>= 1) { if (c < s) sm[c] += sm[c + s]; __syncthreads(); }
    float w = e / sm[0];
    shift[bh * 64 + c] = (int)w;   // truncation: 0, or 1 iff w rounds to 1.0f
}

// agg[b,t,h,c] = vp[b, (t - shift) mod L, h, c]   (bf16 copy)
__global__ void k_gather(const unsigned short* __restrict__ vp,
                         const int* __restrict__ shift,
                         unsigned short* __restrict__ agg) {
    int idx = blockIdx.x * blockDim.x + threadIdx.x;  // over MROWS*DMODEL
    int d  = idx & (DMODEL - 1);
    int bt = idx >> 10;
    int t  = bt & (LSEQ - 1);
    int b  = bt >> 11;
    int s  = shift[b * DMODEL + d];
    int ts = (t - s) & (LSEQ - 1);
    agg[idx] = vp[((size_t)(b * LSEQ + ts)) * DMODEL + d];
}

// ---------------------------------------------------------------------------
extern "C" void kernel_launch(void* const* d_in, const int* in_sizes, int n_in,
                              void* d_out, int out_size, void* d_ws, size_t ws_size,
                              hipStream_t stream) {
    (void)in_sizes; (void)n_in; (void)out_size; (void)ws_size;
    const float* q  = (const float*)d_in[0];
    const float* k  = (const float*)d_in[1];
    const float* v  = (const float*)d_in[2];
    const float* Wq = (const float*)d_in[3];
    const float* bq = (const float*)d_in[4];
    const float* Wk = (const float*)d_in[5];
    const float* bk = (const float*)d_in[6];
    const float* Wv = (const float*)d_in[7];
    const float* bv = (const float*)d_in[8];
    const float* Wo = (const float*)d_in[9];
    const float* bo = (const float*)d_in[10];
    float* out = (float*)d_out;

    // ---- workspace layout (~150 MB) ----
    char* ws = (char*)d_ws; size_t off = 0;
    auto alloc = [&](size_t bytes) -> char* {
        char* p = ws + off; off += (bytes + 255) & ~(size_t)255; return p;
    };
    unsigned short* wq  = (unsigned short*)alloc((size_t)DMODEL * DMODEL * 2);
    unsigned short* wk  = (unsigned short*)alloc((size_t)DMODEL * DMODEL * 2);
    unsigned short* wv  = (unsigned short*)alloc((size_t)DMODEL * DMODEL * 2);
    unsigned short* wo  = (unsigned short*)alloc((size_t)DMODEL * DMODEL * 2);
    unsigned short* Bf  = (unsigned short*)alloc((size_t)LSEQ * NF2 * 2);
    unsigned short* Bi  = (unsigned short*)alloc((size_t)NF2 * LSEQ * 2);
    unsigned short* qpT = (unsigned short*)alloc((size_t)NCH * LSEQ * 2);
    unsigned short* kpT = (unsigned short*)alloc((size_t)NCH * LSEQ * 2);
    unsigned short* vp  = (unsigned short*)alloc((size_t)MROWS * DMODEL * 2);
    unsigned short* Fq  = (unsigned short*)alloc((size_t)NCH * NF2 * 2);
    unsigned short* Fk  = (unsigned short*)alloc((size_t)NCH * NF2 * 2);
    unsigned short* P   = (unsigned short*)alloc((size_t)NCH * NF2 * 2);
    float*          mb  = (float*)alloc((size_t)NCH * 4);
    int*            sh  = (int*)alloc((size_t)NCH * 4);
    unsigned short* agg = (unsigned short*)alloc((size_t)MROWS * DMODEL * 2);

    const int T = 256;
    // ---- prep: weights -> bf16, DFT bases ----
    int nw = DMODEL * DMODEL;
    k_f32_to_bf16<<<(nw + T - 1) / T, T, 0, stream>>>(Wq, wq, nw);
    k_f32_to_bf16<<<(nw + T - 1) / T, T, 0, stream>>>(Wk, wk, nw);
    k_f32_to_bf16<<<(nw + T - 1) / T, T, 0, stream>>>(Wv, wv, nw);
    k_f32_to_bf16<<<(nw + T - 1) / T, T, 0, stream>>>(Wo, wo, nw);
    k_basis_fwd<<<(LSEQ * NF2) / T, T, 0, stream>>>(Bf);
    k_basis_inv<<<(NF2 * LSEQ) / T, T, 0, stream>>>(Bi);

    // ---- projections (fp32 A, bias) ----
    dim3 gp(DMODEL / 128, MROWS / 128);
    k_gemm<0, 2, true><<<gp, T, 0, stream>>>(q, wq, bq, qpT, MROWS, DMODEL, DMODEL);
    k_gemm<0, 2, true><<<gp, T, 0, stream>>>(k, wk, bk, kpT, MROWS, DMODEL, DMODEL);
    k_gemm<0, 1, true><<<gp, T, 0, stream>>>(v, wv, bv, vp,  MROWS, DMODEL, DMODEL);

    // ---- forward DFT (rfft as GEMM, async-LDS A staging) ----
    dim3 gf(NF2 / 128, NCH / 128);
    k_gemm<1, 1, false><<<gf, T, 0, stream>>>(qpT, Bf, nullptr, Fq, NCH, NF2, LSEQ);
    k_gemm<1, 1, false><<<gf, T, 0, stream>>>(kpT, Bf, nullptr, Fk, NCH, NF2, LSEQ);

    // ---- spectral product & inverse DFT with fused max-over-lag ----
    k_spectrum<<<((size_t)NCH * NF2) / T, T, 0, stream>>>(Fq, Fk, P);
    k_init_max<<<(NCH + T - 1) / T, T, 0, stream>>>(mb, NCH);
    dim3 gi(LSEQ / 128, NCH / 128);
    k_gemm<1, 3, false><<<gi, T, 0, stream>>>(P, Bi, nullptr, mb, NCH, LSEQ, NF2);

    // ---- softmax->shift, roll-gather, output projection ----
    k_shift<<<BATCH * NHEAD, NDK, 0, stream>>>(mb, sh);
    k_gather<<<((size_t)MROWS * DMODEL) / T, T, 0, stream>>>(vp, sh, agg);
    k_gemm<1, 0, true><<<gp, T, 0, stream>>>(agg, wo, bo, out, MROWS, DMODEL, DMODEL);
}
